// Model_7816840479070
// MI455X (gfx1250) — matrix operand order repaired and checked by reading, not compiled, not run on hardware
//
#include <hip/hip_runtime.h>
#include <hip/hip_bf16.h>
#include <math.h>
#include <stdint.h>

// ---------------------------------------------------------------------------
// CDNA5 WMMA (wave32): v_wmma_f32_16x16x32_bf16
// ---------------------------------------------------------------------------
typedef __bf16 v16bf __attribute__((ext_vector_type(16)));
typedef float  v8f   __attribute__((ext_vector_type(8)));

#define D_MODEL 768
#define NHEAD   12
#define DHEAD   64
#define FFDIM   3072
#define BATCH   128

// ---------------------------------------------------------------------------
// gfx1250 async global->LDS copy (ASYNCcnt tracked) + fence.
// ---------------------------------------------------------------------------
__device__ __forceinline__ unsigned lds_off(const void* p) {
  return (unsigned)(uintptr_t)p;
}
__device__ __forceinline__ void async_copy_b128(unsigned dst_lds, const void* g) {
  asm volatile("global_load_async_to_lds_b128 %0, %1, off"
               :: "v"(dst_lds), "v"((unsigned long long)(uintptr_t)g)
               : "memory");
}
__device__ __forceinline__ void wait_async0() {
  asm volatile("s_wait_asynccnt 0x0" ::: "memory");
}

// ---------------------------------------------------------------------------
// Fragment builders (contiguous k order: k = 16*half + s, same for A and B)
// ---------------------------------------------------------------------------
__device__ __forceinline__ v16bf frag_a_bf(const __bf16* base, int stride,
                                           int row, int k0, int lane) {
  v16bf a;
  int half = lane >> 4;
  const __bf16* p = base + row * stride + k0;
#pragma unroll
  for (int v = 0; v < 8; ++v) {
    int kb = 2 * v + 16 * half;
    a[2 * v]     = p[kb];
    a[2 * v + 1] = p[kb + 1];
  }
  return a;
}

__device__ __forceinline__ v16bf frag_a_f32(const float* base, int stride,
                                            int row, int k0, int lane) {
  v16bf a;
  int half = lane >> 4;
  const float* p = base + row * stride + k0;
#pragma unroll
  for (int v = 0; v < 8; ++v) {
    int kb = 2 * v + 16 * half;
    a[2 * v]     = (__bf16)p[kb];
    a[2 * v + 1] = (__bf16)p[kb + 1];
  }
  return a;
}

// B fragment from [n][k]-major LDS tile: 16 contiguous bf16 -> 2 aligned b128
__device__ __forceinline__ v16bf frag_b_nk(const __bf16* base, int stride,
                                           int n, int k0, int lane) {
  const __bf16* p = base + n * stride + k0 + 16 * (lane >> 4);
  union { uint4 u[2]; v16bf v; } r;
  r.u[0] = *(const uint4*)(p);
  r.u[1] = *(const uint4*)(p + 8);
  return r.v;
}

// B fragment from [k][n]-major LDS tile (P@V)
__device__ __forceinline__ v16bf frag_b_kn(const __bf16* base, int stride,
                                           int n, int k0, int lane) {
  v16bf b;
  int kh = lane >> 4;
#pragma unroll
  for (int v = 0; v < 8; ++v) {
    int k = k0 + 16 * kh + 2 * v;
    b[2 * v]     = base[k * stride + n];
    b[2 * v + 1] = base[(k + 1) * stride + n];
  }
  return b;
}

__device__ __forceinline__ float gelu_exact(float x) {
  return 0.5f * x * (1.0f + erff(x * 0.70710678118654752f));
}

// ---------------------------------------------------------------------------
// BN on CLS token
// ---------------------------------------------------------------------------
__global__ __launch_bounds__(256) void bn_kernel(
    const float* __restrict__ feats, int L,
    const float* __restrict__ g, const float* __restrict__ bb,
    const float* __restrict__ m, const float* __restrict__ v,
    float* __restrict__ out) {
  int n = blockIdx.x * 256 + threadIdx.x;
  if (n >= BATCH * D_MODEL) return;
  int b = n / D_MODEL, d = n - b * D_MODEL;
  float x = feats[(size_t)b * L * D_MODEL + d];
  out[n] = (x - m[d]) * rsqrtf(v[d] + 1e-5f) * g[d] + bb[d];
}

// ---------------------------------------------------------------------------
// top-2K selection (score[:,0] zeroed), stable tie-break (lowest idx)
// ---------------------------------------------------------------------------
__global__ __launch_bounds__(256) void topk_kernel(
    const float* __restrict__ score, int L, int topn, int* __restrict__ idxout) {
  int b = blockIdx.x, t = threadIdx.x;
  __shared__ float s[224];
  __shared__ float rv[256];
  __shared__ int   ri[256];
  if (t < L) s[t] = (t == 0) ? 0.f : score[b * L + t];
  __syncthreads();
  for (int j = 0; j < topn; ++j) {
    rv[t] = (t < L) ? s[t] : -1e30f;
    ri[t] = t;
    __syncthreads();
    for (int o = 128; o; o >>= 1) {
      if (t < o) {
        if (rv[t + o] > rv[t] || (rv[t + o] == rv[t] && ri[t + o] < ri[t])) {
          rv[t] = rv[t + o]; ri[t] = ri[t + o];
        }
      }
      __syncthreads();
    }
    if (t == 0) { idxout[b * topn + j] = ri[0]; s[ri[0]] = -1e30f; }
    __syncthreads();
  }
}

// ---------------------------------------------------------------------------
// LayerNorm over 768, f32 -> bf16
// ---------------------------------------------------------------------------
__global__ __launch_bounds__(256) void ln_kernel(
    const float* __restrict__ x, const float* __restrict__ g,
    const float* __restrict__ bb, __bf16* __restrict__ out) {
  int row = blockIdx.x, t = threadIdx.x;
  const float* p = x + (size_t)row * D_MODEL;
  float v0 = p[t], v1 = p[t + 256], v2 = p[t + 512];
  __shared__ float red[256];
  red[t] = v0 + v1 + v2;
  __syncthreads();
  for (int o = 128; o; o >>= 1) { if (t < o) red[t] += red[t + o]; __syncthreads(); }
  float mu = red[0] * (1.f / 768.f);
  __syncthreads();
  float d0 = v0 - mu, d1 = v1 - mu, d2 = v2 - mu;
  red[t] = d0 * d0 + d1 * d1 + d2 * d2;
  __syncthreads();
  for (int o = 128; o; o >>= 1) { if (t < o) red[t] += red[t + o]; __syncthreads(); }
  float rstd = rsqrtf(red[0] * (1.f / 768.f) + 1e-5f);
  __bf16* q = out + (size_t)row * D_MODEL;
  q[t]       = (__bf16)(d0 * rstd * g[t] + bb[t]);
  q[t + 256] = (__bf16)(d1 * rstd * g[t + 256] + bb[t + 256]);
  q[t + 512] = (__bf16)(d2 * rstd * g[t + 512] + bb[t + 512]);
}

// ---------------------------------------------------------------------------
// Gather selected tokens (+CLS) and LayerNorm; raw f32 kept for residual
// ---------------------------------------------------------------------------
__global__ __launch_bounds__(256) void gather_ln_kernel(
    const float* __restrict__ feats, const int* __restrict__ idx,
    int L, int Lq, int Kp, int part,
    const float* __restrict__ g, const float* __restrict__ bb,
    float* __restrict__ xsel, __bf16* __restrict__ qn) {
  int row = blockIdx.x, t = threadIdx.x;
  int b = row / Lq, l = row - b * Lq;
  int src = (l == 0) ? 0 : idx[b * (2 * Kp) + part * Kp + (l - 1)];
  const float* p = feats + ((size_t)b * L + src) * D_MODEL;
  float v0 = p[t], v1 = p[t + 256], v2 = p[t + 512];
  float* xo = xsel + (size_t)row * D_MODEL;
  xo[t] = v0; xo[t + 256] = v1; xo[t + 512] = v2;
  __shared__ float red[256];
  red[t] = v0 + v1 + v2;
  __syncthreads();
  for (int o = 128; o; o >>= 1) { if (t < o) red[t] += red[t + o]; __syncthreads(); }
  float mu = red[0] * (1.f / 768.f);
  __syncthreads();
  float d0 = v0 - mu, d1 = v1 - mu, d2 = v2 - mu;
  red[t] = d0 * d0 + d1 * d1 + d2 * d2;
  __syncthreads();
  for (int o = 128; o; o >>= 1) { if (t < o) red[t] += red[t + o]; __syncthreads(); }
  float rstd = rsqrtf(red[0] * (1.f / 768.f) + 1e-5f);
  __bf16* q = qn + (size_t)row * D_MODEL;
  q[t]       = (__bf16)(d0 * rstd * g[t] + bb[t]);
  q[t + 256] = (__bf16)(d1 * rstd * g[t + 256] + bb[t + 256]);
  q[t + 512] = (__bf16)(d2 * rstd * g[t + 512] + bb[t + 512]);
}

// ---------------------------------------------------------------------------
// WMMA GEMM: C[M,N] = act(A_bf16[M,K] @ W_f32[K,N] + bias) (+res)
// 128x64x32 tile, 8 waves; double-buffered LDS, ONE barrier per k-step:
//   store B(i); wait_async(A(i)); barrier; prefetch A/W(i+1); 4x wmma on (i)
// ---------------------------------------------------------------------------
template <bool GELU, bool RES, bool F32OUT>
__global__ __launch_bounds__(256) void gemm_bf16_kernel(
    const __bf16* __restrict__ A, const float* __restrict__ W,
    const float* __restrict__ bias, const float* __restrict__ res,
    float* __restrict__ outF, __bf16* __restrict__ outH,
    int M, int N, int K) {
  __shared__ __bf16 As[2][128][40];   // [m][k], 80B rows
  __shared__ __bf16 Bs[2][64][40];    // [n][k]
  int m0 = blockIdx.y * 128, n0 = blockIdx.x * 64;
  int t = threadIdx.x, lane = t & 31, w = t >> 5;

  // A staging: thread -> (row, 16-elem chunk)
  int arow = t >> 1, acg = (t & 1) * 16;
  int gmA = m0 + arow; if (gmA > M - 1) gmA = M - 1;
  const __bf16* aptr = A + (size_t)gmA * K + acg;
  unsigned aoff0 = lds_off(&As[0][arow][acg]);
  unsigned aoff1 = lds_off(&As[1][arow][acg]);

  // B staging: thread -> (n, k-octet); coalesced along n
  int bn = t & 63, bkb = (t >> 6) * 8;
  const float* wptr = W + (size_t)bkb * N + n0 + bn;

  int frow = w * 16 + (lane & 15);
  v8f acc[4] = {};

  // prologue: prefetch tile 0
  async_copy_b128(aoff0, aptr);
  async_copy_b128(aoff0 + 16, aptr + 8);
  float wreg[8];
#pragma unroll
  for (int i = 0; i < 8; ++i) wreg[i] = wptr[(size_t)i * N];

  int ib = 0;
  for (int kk = 0; kk < K; kk += 32, ib ^= 1) {
    // stage B tile (data loaded last iteration / prologue)
    union { uint4 u; __bf16 h[8]; } bu;
#pragma unroll
    for (int i = 0; i < 8; ++i) bu.h[i] = (__bf16)wreg[i];
    *(uint4*)&Bs[ib][bn][bkb] = bu.u;
    wait_async0();                 // this tile's async A copies arrived
    __syncthreads();               // everyone done with compute(i-1)
    if (kk + 32 < K) {             // prefetch next tile into other buffer
      unsigned ao = ib ? aoff0 : aoff1;
      async_copy_b128(ao, aptr + kk + 32);
      async_copy_b128(ao + 16, aptr + kk + 40);
#pragma unroll
      for (int i = 0; i < 8; ++i) wreg[i] = wptr[(size_t)(kk + 32 + i) * N];
    }
    v16bf a = frag_a_bf(&As[ib][0][0], 40, frow, 0, lane);
#pragma unroll
    for (int c = 0; c < 4; ++c) {
      v16bf b = frag_b_nk(&Bs[ib][0][0], 40, c * 16 + (lane & 15), 0, lane);
      acc[c] = __builtin_amdgcn_wmma_f32_16x16x32_bf16(false, a, false, b,
                                                       (short)0, acc[c], false, false);
    }
  }

  int half = lane >> 4, nl = lane & 15;
#pragma unroll
  for (int r = 0; r < 8; ++r) {
    int gm = m0 + w * 16 + r + 8 * half;
    if (gm >= M) continue;
#pragma unroll
    for (int c = 0; c < 4; ++c) {
      int gn = n0 + c * 16 + nl;
      float vv = acc[c][r] + bias[gn];
      if (GELU) vv = gelu_exact(vv);
      size_t o = (size_t)gm * N + gn;
      if (RES) vv += res[o];
      if (F32OUT) outF[o] = vv;
      else        outH[o] = (__bf16)vv;
    }
  }
}

// ---------------------------------------------------------------------------
// WMMA attention: block (4 waves) per (b, h, 16-query tile).
// Double-buffered async k/v chunks; padded score cols -> -1e30 (weight 0).
// Key-mask token is loaded ONCE per lane per chunk, before the WMMAs.
// ---------------------------------------------------------------------------
__global__ __launch_bounds__(128) void attn_kernel(
    const __bf16* __restrict__ q, const __bf16* __restrict__ k,
    const __bf16* __restrict__ v, const int* __restrict__ tokens,
    __bf16* __restrict__ out, int Lq, int Lk, int nc) {
  __shared__ __bf16 qs[16][72];
  __shared__ float  sc[16][224];
  __shared__ __bf16 kvs[2][32][72];
  int qb = blockIdx.x, h = blockIdx.y, b = blockIdx.z;
  int t = threadIdx.x, lane = t & 31, w = t >> 5;
  int width = nc * 32;

  // kv staging indices: 32x64, 16 bf16 per thread
  int r2 = t >> 2, cg2 = (t & 3) * 16;
  unsigned kvoff0 = lds_off(&kvs[0][r2][cg2]);
  unsigned kvoff1 = lds_off(&kvs[1][r2][cg2]);

  {   // q tile: 16x64, 8 bf16 per thread (async)
    int r = t >> 3, cg = (t & 7) * 8;
    int l = qb * 16 + r; if (l > Lq - 1) l = Lq - 1;
    async_copy_b128(lds_off(&qs[r][cg]),
                    q + ((size_t)(b * Lq + l) * D_MODEL + h * DHEAD + cg));
  }
  {   // k chunk 0 (async)
    int kr = r2; if (kr > Lk - 1) kr = Lk - 1;
    const __bf16* src = k + ((size_t)(b * Lk + kr) * D_MODEL + h * DHEAD + cg2);
    async_copy_b128(kvoff0, src);
    async_copy_b128(kvoff0 + 16, src + 8);
  }

  // ---- Pass 1: S = q K^T / 8 (+mask) ----
  for (int c = 0; c < nc; ++c) {
    wait_async0();
    __syncthreads();
    if (c + 1 < nc) {
      int kr = (c + 1) * 32 + r2; if (kr > Lk - 1) kr = Lk - 1;
      const __bf16* src = k + ((size_t)(b * Lk + kr) * D_MODEL + h * DHEAD + cg2);
      unsigned ko = (c & 1) ? kvoff0 : kvoff1;
      async_copy_b128(ko, src);
      async_copy_b128(ko + 16, src + 8);
    }
    if (w < 2) {
      int nl = lane & 15;
      int kcol = c * 32 + w * 16 + nl;
      // hoisted, unconditional (clamped) token load: issues before the WMMAs
      int tok = 1;
      if (tokens) {
        int kidx = (kcol < Lk) ? kcol : (Lk - 1);
        tok = tokens[b * Lk + kidx];
      }
      const __bf16* kb = &kvs[c & 1][0][0];
      v8f acc = {};
      int arow = lane & 15;
#pragma unroll
      for (int ks = 0; ks < 64; ks += 32) {
        v16bf a  = frag_a_bf(&qs[0][0], 72, arow, ks, lane);
        v16bf bm = frag_b_nk(kb, 72, w * 16 + nl, ks, lane);
        acc = __builtin_amdgcn_wmma_f32_16x16x32_bf16(false, a, false, bm,
                                                      (short)0, acc, false, false);
      }
      bool oob = (kcol >= Lk);
      bool msk = (tok == 0) && !oob;
      int half = lane >> 4;
#pragma unroll
      for (int r = 0; r < 8; ++r) {
        int row = r + 8 * half;
        float sv = acc[r] * 0.125f;                 // 1/sqrt(64)
        if (oob) sv = -1e30f;                       // padding -> weight 0
        else if (msk) sv = -1e9f;                   // key mask
        sc[row][c * 32 + w * 16 + nl] = sv;
      }
    }
  }
  __syncthreads();

  // ---- Pass 2: softmax (wave per 4 rows) ----
#pragma unroll
  for (int i = 0; i < 4; ++i) {
    int r = w * 4 + i;
    float m = -1e30f;
    for (int cj = lane; cj < width; cj += 32) m = fmaxf(m, sc[r][cj]);
    for (int o = 16; o; o >>= 1) m = fmaxf(m, __shfl_xor(m, o, 32));
    float s = 0.f;
    for (int cj = lane; cj < width; cj += 32) s += __expf(sc[r][cj] - m);
    for (int o = 16; o; o >>= 1) s += __shfl_xor(s, o, 32);
    float inv = 1.f / fmaxf(s, 1e-30f);
    for (int cj = lane; cj < width; cj += 32)
      sc[r][cj] = __expf(sc[r][cj] - m) * inv;
  }
  __syncthreads();   // sc final; all pass-1 kvs readers retired

  {   // v chunk 0 (async)
    int kr = r2; if (kr > Lk - 1) kr = Lk - 1;
    const __bf16* src = v + ((size_t)(b * Lk + kr) * D_MODEL + h * DHEAD + cg2);
    async_copy_b128(kvoff0, src);
    async_copy_b128(kvoff0 + 16, src + 8);
  }

  // ---- Pass 3: O = P @ V (wave w owns dh columns w*16..) ----
  v8f acc = {};
  for (int c = 0; c < nc; ++c) {
    wait_async0();
    __syncthreads();
    if (c + 1 < nc) {
      int kr = (c + 1) * 32 + r2; if (kr > Lk - 1) kr = Lk - 1;
      const __bf16* src = v + ((size_t)(b * Lk + kr) * D_MODEL + h * DHEAD + cg2);
      unsigned ko = (c & 1) ? kvoff0 : kvoff1;
      async_copy_b128(ko, src);
      async_copy_b128(ko + 16, src + 8);
    }
    v16bf a  = frag_a_f32(&sc[0][0], 224, lane & 15, c * 32, lane);
    v16bf bm = frag_b_kn(&kvs[c & 1][0][0], 72, w * 16 + (lane & 15), 0, lane);
    acc = __builtin_amdgcn_wmma_f32_16x16x32_bf16(false, a, false, bm,
                                                  (short)0, acc, false, false);
  }
  int half = lane >> 4, nl = lane & 15;
#pragma unroll
  for (int r = 0; r < 8; ++r) {
    int row = r + 8 * half;
    int l = qb * 16 + row;
    if (l < Lq)
      out[(size_t)(b * Lq + l) * D_MODEL + h * DHEAD + w * 16 + nl] = (__bf16)acc[r];
  }
}

// ---------------------------------------------------------------------------
// Host side
// ---------------------------------------------------------------------------
static void launch_gemm(const __bf16* A, const float* W, const float* bias,
                        const float* res, float* outF, __bf16* outH,
                        int M, int N, int K, int act, hipStream_t s) {
  dim3 g(N / 64, (M + 127) / 128);
  if (outF)      gemm_bf16_kernel<false, true, true><<<g, 256, 0, s>>>(
                     A, W, bias, res, outF, outH, M, N, K);
  else if (act)  gemm_bf16_kernel<true, false, false><<<g, 256, 0, s>>>(
                     A, W, bias, res, outF, outH, M, N, K);
  else           gemm_bf16_kernel<false, false, false><<<g, 256, 0, s>>>(
                     A, W, bias, res, outF, outH, M, N, K);
}

extern "C" void kernel_launch(void* const* d_in, const int* in_sizes, int n_in,
                              void* d_out, int out_size, void* d_ws, size_t ws_size,
                              hipStream_t stream) {
  const float* image_feats = (const float*)d_in[0];
  const float* image_att   = (const float*)d_in[1];
  const float* text_feats  = (const float*)d_in[2];
  const float* text_att    = (const float*)d_in[3];
  const int*   tokens      = (const int*)d_in[4];
  const float* bn_img_g = (const float*)d_in[5];
  const float* bn_img_b = (const float*)d_in[6];
  const float* bn_img_m = (const float*)d_in[7];
  const float* bn_img_v = (const float*)d_in[8];
  const float* bn_txt_g = (const float*)d_in[9];
  const float* bn_txt_b = (const float*)d_in[10];
  const float* bn_txt_m = (const float*)d_in[11];
  const float* bn_txt_v = (const float*)d_in[12];
  const float* ln1_g = (const float*)d_in[13];
  const float* ln1_b = (const float*)d_in[14];
  const float* ln2_g = (const float*)d_in[15];
  const float* ln2_b = (const float*)d_in[16];
  const float* Wq = (const float*)d_in[17];
  const float* Wk = (const float*)d_in[18];
  const float* Wv = (const float*)d_in[19];
  const float* Wo = (const float*)d_in[20];
  const float* bq = (const float*)d_in[21];
  const float* bk = (const float*)d_in[22];
  const float* bv = (const float*)d_in[23];
  const float* bo = (const float*)d_in[24];
  const float* W1 = (const float*)d_in[25];
  const float* b1 = (const float*)d_in[26];
  const float* W2 = (const float*)d_in[27];
  const float* b2 = (const float*)d_in[28];
  const int IMG_K = 59, TXT_K = 19;   // reference scalars

  float* out = (float*)d_out;
  float* out_img_glob = out;                       // 128*768
  float* out_txt_glob = out + 98304;
  float* out_ip1 = out + 196608;                   // 128*60*768
  float* out_ip2 = out + 6094848;
  float* out_tp1 = out + 11993088;                 // 128*20*768
  float* out_tp2 = out + 13959168;

  char* wsp = (char*)d_ws;
  auto carve = [&](size_t bytes) -> void* {
    void* p = (void*)wsp;
    wsp += (bytes + 255) & ~(size_t)255;
    return p;
  };
  int* idx_img = (int*)carve((size_t)BATCH * 118 * 4);
  int* idx_txt = (int*)carve((size_t)BATCH * 38 * 4);
  __bf16* s_knff = (__bf16*)carve((size_t)23592960 * 2);  // kn | ff (disjoint lifetimes)
  __bf16* s_k    = (__bf16*)carve((size_t)19365888 * 2);
  __bf16* s_v    = (__bf16*)carve((size_t)19365888 * 2);
  float*  s_xsel = (float*)carve((size_t)5898240 * 4);
  __bf16* s_qnh  = (__bf16*)carve((size_t)5898240 * 2);   // qn then ln2(h)
  __bf16* s_qatt = (__bf16*)carve((size_t)5898240 * 2);   // q then att

  bn_kernel<<<dim3((BATCH * D_MODEL + 255) / 256), 256, 0, stream>>>(
      image_feats, 197, bn_img_g, bn_img_b, bn_img_m, bn_img_v, out_img_glob);
  bn_kernel<<<dim3((BATCH * D_MODEL + 255) / 256), 256, 0, stream>>>(
      text_feats, 64, bn_txt_g, bn_txt_b, bn_txt_m, bn_txt_v, out_txt_glob);

  topk_kernel<<<BATCH, 256, 0, stream>>>(image_att, 197, 2 * IMG_K, idx_img);
  topk_kernel<<<BATCH, 256, 0, stream>>>(text_att, 64, 2 * TXT_K, idx_txt);

  struct Mod {
    const float* feats; const int* idx; const int* toks;
    int L, Kp, Lq; float* outp[2];
  };
  Mod mods[2] = {
    { image_feats, idx_img, nullptr, 197, IMG_K, IMG_K + 1, { out_ip1, out_ip2 } },
    { text_feats,  idx_txt, tokens,   64, TXT_K, TXT_K + 1, { out_tp1, out_tp2 } },
  };

  for (int mi = 0; mi < 2; ++mi) {
    const Mod& md = mods[mi];
    int Mk = BATCH * md.L;
    int nc = (md.L + 31) / 32;
    ln_kernel<<<Mk, 256, 0, stream>>>(md.feats, ln1_g, ln1_b, s_knff);
    launch_gemm(s_knff, Wk, bk, nullptr, nullptr, s_k, Mk, D_MODEL, D_MODEL, 0, stream);
    launch_gemm(s_knff, Wv, bv, nullptr, nullptr, s_v, Mk, D_MODEL, D_MODEL, 0, stream);

    for (int part = 0; part < 2; ++part) {
      int Mq = BATCH * md.Lq;
      float* op = md.outp[part];
      gather_ln_kernel<<<Mq, 256, 0, stream>>>(md.feats, md.idx, md.L, md.Lq,
                                               md.Kp, part, ln1_g, ln1_b,
                                               s_xsel, s_qnh);
      launch_gemm(s_qnh, Wq, bq, nullptr, nullptr, s_qatt, Mq, D_MODEL, D_MODEL, 0, stream);
      attn_kernel<<<dim3((md.Lq + 15) / 16, NHEAD, BATCH), 128, 0, stream>>>(
          s_qatt, s_k, s_v, md.toks, s_qatt, md.Lq, md.L, nc);
      launch_gemm(s_qatt, Wo, bo, s_xsel, op, nullptr, Mq, D_MODEL, D_MODEL, 0, stream);
      ln_kernel<<<Mq, 256, 0, stream>>>(op, ln2_g, ln2_b, s_qnh);
      launch_gemm(s_qnh, W1, b1, nullptr, nullptr, s_knff, Mq, FFDIM, D_MODEL, 1, stream);
      launch_gemm(s_knff, W2, b2, op, op, nullptr, Mq, D_MODEL, FFDIM, 0, stream);
    }
  }
}
